// Model_26207890440933
// MI455X (gfx1250) — matrix operand order repaired and checked by reading, not compiled, not run on hardware
//
#include <hip/hip_runtime.h>
#include <hip/hip_bf16.h>

typedef __attribute__((ext_vector_type(16))) _Float16 v16h;
typedef __attribute__((ext_vector_type(8)))  float    v8f;
typedef __attribute__((ext_vector_type(2)))  float    v2f;

#define B_   64
#define T_   128
#define F_   9
#define H_   16
#define G_   64      // 4*H
#define IN0_ 65
#define V_   32000
#define BT_  (B_*T_) // 8192

// Fast device math: raw v_rcp_f32 / v_exp_f32, no IEEE div fixup chains.
__device__ __forceinline__ float rcpf(float x)  { return __builtin_amdgcn_rcpf(x); }
__device__ __forceinline__ float sigf(float x)  { return rcpf(1.0f + __expf(-x)); }
__device__ __forceinline__ float tanhff(float x){ return 1.0f - 2.0f * rcpf(1.0f + __expf(2.0f * x)); }

// ---------------------------------------------------------------------------
// Kernel 1: embedding gather + layer0 input projection (adds b_ih0+b_hh0).
// grid = BT_ blocks x 64 threads. Output xp0 in [t][b][g] layout so the
// recurrence kernel reads one contiguous 4KB block per step.
// ---------------------------------------------------------------------------
__global__ void embed_proj_kernel(const float* __restrict__ x,
                                  const float* __restrict__ emb,
                                  const float* __restrict__ Wih0,
                                  const float* __restrict__ bih0,
                                  const float* __restrict__ bhh0,
                                  float* __restrict__ xp0) {
  __shared__ float sin[IN0_];
  const int bt  = blockIdx.x;
  const int b   = bt >> 7;                   // bt / T_
  const int t   = bt & (T_ - 1);
  const int tid = threadIdx.x;               // 0..63 (gate index g)
  const float* xr = x + bt * F_;
  {
    int j = tid >> 3, k = tid & 7;
    int idx = (int)xr[j];                    // cat index stored as float
    sin[tid] = emb[idx * 8 + k];
  }
  if (tid == 0) sin[64] = xr[8];             // numeric feature
  __syncthreads();
  const float* wr = Wih0 + tid * IN0_;
  float acc = bih0[tid] + bhh0[tid];
  #pragma unroll
  for (int i = 0; i < IN0_; ++i) acc = fmaf(sin[i], wr[i], acc);
  xp0[((size_t)t * B_ + b) * G_ + tid] = acc;
}

// ---------------------------------------------------------------------------
// Kernel 2: fused 2-layer LSTM recurrence. grid = 4 blocks x 32 threads.
// ONE wave owns 16 batch rows and all four gate N-tiles: four independent
// v_wmma_f32_16x16x4_f32 accumulator chains (good XDL ILP), and i/f/g/o for a
// given (batch,hidden) land in the same lane/VGPR slot -> the whole gate
// nonlinearity is straight-line register code, no LDS gate exchange.
// Only h needs a wave-private LDS tile to re-shape C-layout -> A-layout.
// Barriers in a single-wave workgroup are S_NOPs (ISA 3.1).
// ---------------------------------------------------------------------------
__global__ void __launch_bounds__(32)
lstm_fused_kernel(const float* __restrict__ xp0,
                  const float* __restrict__ h0in,
                  const float* __restrict__ c0in,
                  const float* __restrict__ Whh0,
                  const float* __restrict__ Wih1,
                  const float* __restrict__ Whh1,
                  const float* __restrict__ bih1,
                  const float* __restrict__ bhh1,
                  float* __restrict__ y1,
                  float* __restrict__ out) {
  __shared__ float sT0[16][H_];   // h0 / y0 tile in row-major (A source)
  __shared__ float sT1[16][H_];   // h1 tile in row-major (A source)

  const int lane  = threadIdx.x & 31;
  const int n     = lane & 15;                 // N col / A row
  const int hiK   = (lane >= 16) ? 2 : 0;
  const int mrow0 = (lane < 16) ? 0 : 8;       // C-layout row base for this lane
  const int bBase = blockIdx.x * 16;

  // Weight B-fragments, register-resident for all 128 steps.
  // B[k][nn] = W[gate = ni*16+nn][k]
  v2f bW0[4][4], bI1[4][4], bH1[4][4];
  #pragma unroll
  for (int ni = 0; ni < 4; ++ni) {
    const int g = ni * 16 + n;
    #pragma unroll
    for (int kk = 0; kk < 4; ++kk) {
      bW0[ni][kk][0] = Whh0[g * H_ + kk * 4 + hiK];
      bW0[ni][kk][1] = Whh0[g * H_ + kk * 4 + hiK + 1];
      bI1[ni][kk][0] = Wih1[g * H_ + kk * 4 + hiK];
      bI1[ni][kk][1] = Wih1[g * H_ + kk * 4 + hiK + 1];
      bH1[ni][kk][0] = Whh1[g * H_ + kk * 4 + hiK];
      bH1[ni][kk][1] = Whh1[g * H_ + kk * 4 + hiK + 1];
    }
  }
  float bias1[4];
  #pragma unroll
  for (int ni = 0; ni < 4; ++ni) bias1[ni] = bih1[ni * 16 + n] + bhh1[ni * 16 + n];

  // Cell state in C layout; h state seeded into the LDS transpose tiles.
  v8f cs0, cs1;
  #pragma unroll
  for (int r = 0; r < 8; ++r) {
    const int mrow = mrow0 + r;
    cs0[r] = c0in[(bBase + mrow) * H_ + n];
    cs1[r] = c0in[B_ * H_ + (bBase + mrow) * H_ + n];
    sT0[mrow][n] = h0in[(bBase + mrow) * H_ + n];
    sT1[mrow][n] = h0in[B_ * H_ + (bBase + mrow) * H_ + n];
  }
  __syncthreads();

  for (int t = 0; t < T_; ++t) {
    const float* xpt = xp0 + ((size_t)t * B_ + bBase) * G_;   // 16x64 contiguous
    if (t + 1 < T_) {                                         // hide next step's latency
      const float* nxt = xp0 + ((size_t)(t + 1) * B_ + bBase) * G_;
      __builtin_prefetch(nxt + lane * 32, 0, 0);              // 32 lanes x 128B = 4KB
    }

    // ---- layer 0: gates = xp + h0 @ Whh0^T ----
    v8f g0[4];
    #pragma unroll
    for (int ni = 0; ni < 4; ++ni)
      #pragma unroll
      for (int r = 0; r < 8; ++r)
        g0[ni][r] = xpt[(mrow0 + r) * G_ + ni * 16 + n];
    v2f a[4];
    #pragma unroll
    for (int kk = 0; kk < 4; ++kk) {
      a[kk][0] = sT0[n][kk * 4 + hiK];
      a[kk][1] = sT0[n][kk * 4 + hiK + 1];
    }
    #pragma unroll
    for (int ni = 0; ni < 4; ++ni)
      #pragma unroll
      for (int kk = 0; kk < 4; ++kk)
        g0[ni] = __builtin_amdgcn_wmma_f32_16x16x4_f32(false, a[kk], false, bW0[ni][kk],
                                                       (short)0, g0[ni], false, false);
    // gate nonlinearity entirely in registers (same lane/slot across gates)
    v8f hv0;
    #pragma unroll
    for (int r = 0; r < 8; ++r) {
      const float ig = sigf(g0[0][r]);
      const float fg = sigf(g0[1][r]);
      const float gg = tanhff(g0[2][r]);
      const float og = sigf(g0[3][r]);
      const float cc = fg * cs0[r] + ig * gg;
      cs0[r] = cc;
      hv0[r] = og * tanhff(cc);
    }
    __syncthreads();                       // order old sT0 reads vs new writes
    #pragma unroll
    for (int r = 0; r < 8; ++r) sT0[mrow0 + r][n] = hv0[r];
    __syncthreads();

    // ---- layer 1: gates = y0 @ Wih1^T + h1 @ Whh1^T + bias ----
    v8f g1[4];
    #pragma unroll
    for (int ni = 0; ni < 4; ++ni)
      #pragma unroll
      for (int r = 0; r < 8; ++r) g1[ni][r] = bias1[ni];
    v2f ay[4], ah[4];
    #pragma unroll
    for (int kk = 0; kk < 4; ++kk) {
      ay[kk][0] = sT0[n][kk * 4 + hiK];  ay[kk][1] = sT0[n][kk * 4 + hiK + 1];
      ah[kk][0] = sT1[n][kk * 4 + hiK];  ah[kk][1] = sT1[n][kk * 4 + hiK + 1];
    }
    #pragma unroll
    for (int ni = 0; ni < 4; ++ni) {
      #pragma unroll
      for (int kk = 0; kk < 4; ++kk) {
        g1[ni] = __builtin_amdgcn_wmma_f32_16x16x4_f32(false, ay[kk], false, bI1[ni][kk],
                                                       (short)0, g1[ni], false, false);
        g1[ni] = __builtin_amdgcn_wmma_f32_16x16x4_f32(false, ah[kk], false, bH1[ni][kk],
                                                       (short)0, g1[ni], false, false);
      }
    }
    v8f hv1;
    #pragma unroll
    for (int r = 0; r < 8; ++r) {
      const float ig = sigf(g1[0][r]);
      const float fg = sigf(g1[1][r]);
      const float gg = tanhff(g1[2][r]);
      const float og = sigf(g1[3][r]);
      const float cc = fg * cs1[r] + ig * gg;
      cs1[r] = cc;
      hv1[r] = og * tanhff(cc);
    }
    __syncthreads();
    #pragma unroll
    for (int r = 0; r < 8; ++r) {
      sT1[mrow0 + r][n] = hv1[r];
      y1[((size_t)(bBase + mrow0 + r) * T_ + t) * H_ + n] = hv1[r];
    }
    __syncthreads();
  }

  // final states -> d_out tail (logits | hN(2,B,H) | cN(2,B,H))
  const size_t HN = (size_t)BT_ * V_;
  const size_t CN = HN + (size_t)2 * B_ * H_;
  #pragma unroll
  for (int r = 0; r < 8; ++r) {
    const int gb = bBase + mrow0 + r;
    out[HN + gb * H_ + n]           = sT0[mrow0 + r][n];
    out[HN + B_ * H_ + gb * H_ + n] = sT1[mrow0 + r][n];
    out[CN + gb * H_ + n]           = cs0[r];
    out[CN + B_ * H_ + gb * H_ + n] = cs1[r];
  }
}

// ---------------------------------------------------------------------------
// Kernel 3: FC head, store-bandwidth bound (1.05 GB of logits, ~45us floor at
// 23.3 TB/s). One v_wmma_f32_16x16x32_f16 per 16x16 tile (K=16 zero-padded),
// B-frags register-resident over 16 M-tiles, nontemporal stores keep the
// 2MB fc_w + 512KB y1 resident in L2.
// ---------------------------------------------------------------------------
__global__ void __launch_bounds__(256)
fc_kernel(const float* __restrict__ y1,
          const float* __restrict__ fcw,
          const float* __restrict__ fcb,
          float* __restrict__ out) {
  const int tid  = threadIdx.x;
  const int wave = tid >> 5;
  const int lane = tid & 31;
  const int n    = lane & 15;
  const int ntile   = blockIdx.x * 8 + wave;  // 0..1999
  const int colBase = ntile * 16;

  // B fragment: lanes 0-15 hold K=0..15 for col N=lane; lanes 16-31 (K=16..31) = 0
  v16h bf = {};
  if (lane < 16) {
    const float* wr = fcw + (size_t)(colBase + lane) * H_;
    #pragma unroll
    for (int k = 0; k < 16; ++k) bf[k] = (_Float16)wr[k];
  }
  const float bias = fcb[colBase + n];

  #pragma unroll 1
  for (int m = 0; m < 16; ++m) {
    const int rowBase = blockIdx.y * 256 + m * 16;
    // A fragment: lane<16 holds K=0..15, lanes 16-31 (K=16..31) = 0
    v16h af = {};
    const float* yr = y1 + (size_t)(rowBase + n) * H_;
    if (lane < 16) {
      #pragma unroll
      for (int k = 0; k < 16; ++k) af[k] = (_Float16)yr[k];
    }
    v8f c = { bias, bias, bias, bias, bias, bias, bias, bias };
    c = __builtin_amdgcn_wmma_f32_16x16x32_f16(false, af, false, bf,
                                               (short)0, c, false, false);
    #pragma unroll
    for (int r = 0; r < 8; ++r) {
      int row = rowBase + ((lane < 16) ? r : (r + 8));
      __builtin_nontemporal_store(c[r], out + (size_t)row * V_ + colBase + n);
    }
  }
}

// ---------------------------------------------------------------------------
extern "C" void kernel_launch(void* const* d_in, const int* in_sizes, int n_in,
                              void* d_out, int out_size, void* d_ws, size_t ws_size,
                              hipStream_t stream) {
  const float* x    = (const float*)d_in[0];
  const float* h0   = (const float*)d_in[1];
  const float* c0   = (const float*)d_in[2];
  const float* emb  = (const float*)d_in[3];
  const float* Wih0 = (const float*)d_in[4];
  const float* Whh0 = (const float*)d_in[5];
  const float* bih0 = (const float*)d_in[6];
  const float* bhh0 = (const float*)d_in[7];
  const float* Wih1 = (const float*)d_in[8];
  const float* Whh1 = (const float*)d_in[9];
  const float* bih1 = (const float*)d_in[10];
  const float* bhh1 = (const float*)d_in[11];
  const float* fcw  = (const float*)d_in[12];
  const float* fcb  = (const float*)d_in[13];
  float* out = (float*)d_out;

  float* ws  = (float*)d_ws;
  float* xp0 = ws;                    // BT_*G_ floats (2 MB), [t][b][g]
  float* y1  = ws + (size_t)BT_ * G_; // BT_*H_ floats (512 KB), [b*T+t][h]

  embed_proj_kernel<<<BT_, 64, 0, stream>>>(x, emb, Wih0, bih0, bhh0, xp0);
  lstm_fused_kernel<<<4, 32, 0, stream>>>(xp0, h0, c0, Whh0, Wih1, Whh1,
                                          bih1, bhh1, y1, out);
  fc_kernel<<<dim3(250, 32), 256, 0, stream>>>(y1, fcw, fcb, out);
}
